// PairformerBlock_12103217840570
// MI455X (gfx1250) — matrix operand order repaired and checked by reading, not compiled, not run on hardware
//
#include <hip/hip_runtime.h>
#include <hip/hip_bf16.h>
#include <math.h>

// ---------------------------------------------------------------------------
// Pairformer block for MI455X (gfx1250, wave32).
// All GEMMs use V_WMMA_F32_16X16X32_F16 (f16 in, f32 accumulate).
// Weights are pre-swizzled into WMMA B-fragment order (contiguous 32B/lane).
// ---------------------------------------------------------------------------

#define NSEQ 256
#define CZ   128
#define CS   384
#define NPOS (NSEQ * NSEQ)          // 65536 pair positions

typedef _Float16 v16h __attribute__((ext_vector_type(16)));
typedef float    v8f  __attribute__((ext_vector_type(8)));

#define DEVINL __device__ __forceinline__

DEVINL v8f wmma16(v16h a, v16h b, v8f c) {
  return __builtin_amdgcn_wmma_f32_16x16x32_f16(false, a, false, b,
                                                (short)0, c, false, false);
}

// ----- WMMA fragment loaders (wave32, this code's k order k = 16*(lane/16)+s in A and B, 16x16x32 f16) -----
// A: 16x32 (MxK). lanes 0-15: row M=lane, halves->K 0..15;
//                 lanes 16-31: row M=lane-16, halves->K 16..31.
DEVINL v16h frag_a_rm(const _Float16* src, int ld) {   // src row-major MxK
  int lane = threadIdx.x & 31;
  int grp = lane >> 4, row = lane & 15;
  const _Float16* p = src + row * ld;
  v16h o;
#pragma unroll
  for (int v = 0; v < 8; ++v) {
    int kb = grp * 16 + 2 * v;
    o[2 * v]     = p[kb];
    o[2 * v + 1] = p[kb + 1];
  }
  return o;
}
// B: 32x16 (KxN). lanes 0-15: N=lane, halves = K 0..15;
//                 lanes 16-31: N=lane-16, halves = K 16..31.
DEVINL v16h frag_b_nk(const _Float16* src, int ld) {   // src row-major NxK (B^T)
  int lane = threadIdx.x & 31;
  int n = lane & 15, grp = lane >> 4;
  const _Float16* p = src + n * ld + 16 * grp;
  v16h o;
#pragma unroll
  for (int h = 0; h < 16; ++h) o[h] = p[h];
  return o;
}
// Pre-swizzled weight fragment: one contiguous v16h per lane.
DEVINL v16h frag_bw(const _Float16* swz, int ktiles, int kt, int nt) {
  int lane = threadIdx.x & 31;
  return *(const v16h*)(swz + ((((long)nt * ktiles + kt) * 32 + lane) << 4));
}
// D: 16x16 f32: lane -> col N=lane&15; reg r -> row M = r + 8*(lane>>4).

DEVINL float wred_sum(float v) {
#pragma unroll
  for (int o = 16; o; o >>= 1) v += __shfl_xor(v, o, 32);
  return v;
}
DEVINL float wred_max(float v) {
#pragma unroll
  for (int o = 16; o; o >>= 1) v = fmaxf(v, __shfl_xor(v, o, 32));
  return v;
}
DEVINL float sigm(float x) { return 1.0f / (1.0f + __expf(-x)); }

// LayerNorm P positions of C channels into LDS as f16. One wave per position.
DEVINL void ln_to_lds(const float* base, long posStride, int C, int P,
                      const float* g, const float* b, _Float16* dst) {
  int wid = threadIdx.x >> 5, lane = threadIdx.x & 31, nw = blockDim.x >> 5;
  for (int q = wid; q < P; q += nw) {
    const float* x = base + (long)q * posStride;
    float s = 0.f;
    for (int c = lane; c < C; c += 32) s += x[c];
    float m = wred_sum(s) / (float)C;
    float v = 0.f;
    for (int c = lane; c < C; c += 32) { float d = x[c] - m; v += d * d; }
    v = wred_sum(v) / (float)C;
    float rs = rsqrtf(v + 1e-5f);
    for (int c = lane; c < C; c += 32)
      dst[q * C + c] = (_Float16)((x[c] - m) * rs * g[c] + b[c]);
  }
}

// ---------------------------------------------------------------------------
// weight conversion + swizzle into WMMA B-fragment order.
// src: K x srcN row-major f32; dst: tiles (nt-major, kt-minor) of
// 32 lanes x 16 contiguous halves; N >= srcN is zero-padded.
// ---------------------------------------------------------------------------
__global__ void k_cvt_swz(const float* W, _Float16* dst, int K, int N, int srcN) {
  int total = K * N;
  int ktiles = K >> 5;
  for (int idx = blockIdx.x * blockDim.x + threadIdx.x; idx < total;
       idx += gridDim.x * blockDim.x) {
    int h = idx & 15;
    int lane = (idx >> 4) & 31;
    int t = idx >> 9;
    int nt = t / ktiles, kt = t - nt * ktiles;
    int n = nt * 16 + (lane & 15);
    int k = kt * 32 + (lane >> 4) * 16 + h;
    dst[idx] = (_Float16)((n < srcN) ? W[(long)k * srcN + n] : 0.f);
  }
}

// ---------------------------------------------------------------------------
// tri_mul: projections  a = sig(zn@Wga+bga)*(zn@Wpa+bpa), b likewise.
// readT=0: store a_t,b_t as [c][row][col] (positions walk a row of z).
// readT=1: store as [c][col][row]        (positions walk a column of z).
// Either way, k_trimul_einsum sees row-major A (MxK) and NxK B per channel.
// ---------------------------------------------------------------------------
__global__ void k_trimul_proj(const float* z, const float* ng, const float* nb,
                              const _Float16* Wpa, const float* bpa,
                              const _Float16* Wga, const float* bga,
                              const _Float16* Wpb, const float* bpb,
                              const _Float16* Wgb, const float* bgb,
                              _Float16* a_t, _Float16* b_t, int readT) {
  __shared__ _Float16 zn[32 * CZ];
  __shared__ _Float16 outA[CZ * 32];
  __shared__ _Float16 outB[CZ * 32];
  int p0 = blockIdx.x * 32;
  int rr = p0 >> 8, c0 = p0 & 255;
  const float* base = readT ? z + ((long)c0 * NSEQ + rr) * CZ
                            : z + (long)p0 * CZ;
  long stride = readT ? (long)NSEQ * CZ : CZ;
  ln_to_lds(base, stride, CZ, 32, ng, nb, zn);
  __syncthreads();
  int wid = threadIdx.x >> 5, lane = threadIdx.x & 31;
  int nn = lane & 15, grp = lane >> 4;
  int ct = wid;                       // 8 col tiles over N=128
  for (int rt = 0; rt < 2; ++rt) {
    {
      v8f ap = {}, ag = {};
#pragma unroll
      for (int kk = 0; kk < 4; ++kk) {
        v16h af = frag_a_rm(zn + rt * 16 * CZ + kk * 32, CZ);
        ap = wmma16(af, frag_bw(Wpa, 4, kk, ct), ap);
        ag = wmma16(af, frag_bw(Wga, 4, kk, ct), ag);
      }
#pragma unroll
      for (int r = 0; r < 8; ++r) {
        int q = rt * 16 + r + 8 * grp, ch = ct * 16 + nn;
        outA[ch * 32 + q] =
            (_Float16)(sigm(ag[r] + bga[ch]) * (ap[r] + bpa[ch]));
      }
    }
    {
      v8f bp = {}, bg = {};
#pragma unroll
      for (int kk = 0; kk < 4; ++kk) {
        v16h af = frag_a_rm(zn + rt * 16 * CZ + kk * 32, CZ);
        bp = wmma16(af, frag_bw(Wpb, 4, kk, ct), bp);
        bg = wmma16(af, frag_bw(Wgb, 4, kk, ct), bg);
      }
#pragma unroll
      for (int r = 0; r < 8; ++r) {
        int q = rt * 16 + r + 8 * grp, ch = ct * 16 + nn;
        outB[ch * 32 + q] =
            (_Float16)(sigm(bg[r] + bgb[ch]) * (bp[r] + bpb[ch]));
      }
    }
  }
  __syncthreads();
  for (int idx = threadIdx.x; idx < CZ * 32; idx += blockDim.x) {
    int ch = idx >> 5, q = idx & 31;
    a_t[(long)ch * NPOS + p0 + q] = outA[idx];
    b_t[(long)ch * NPOS + p0 + q] = outB[idx];
  }
}

// Per-channel 256x256x256 GEMMs: U_c = A_c @ B_c^T, A row-major MxK,
// B stored NxK. upd written [i][j][c] f16. (Transposition for the incoming
// variant is baked into the a_t/b_t layouts by k_trimul_proj.)
__global__ void k_trimul_einsum(const _Float16* a_t, const _Float16* b_t,
                                _Float16* upd) {
  int wid = threadIdx.x >> 5, lane = threadIdx.x & 31;
  int nn = lane & 15, grp = lane >> 4;
  int c = blockIdx.z * 8 + wid;
  int i0 = blockIdx.x * 64, j0 = blockIdx.y * 64;
  const _Float16* A = a_t + (long)c * NPOS;
  const _Float16* B = b_t + (long)c * NPOS;
  for (int mt = 0; mt < 4; ++mt)
    for (int nt = 0; nt < 4; ++nt) {
      v8f acc = {};
#pragma unroll
      for (int kk = 0; kk < 8; ++kk) {
        int k = kk * 32;
        v16h af = frag_a_rm(A + (i0 + mt * 16) * NSEQ + k, NSEQ);
        v16h bf = frag_b_nk(B + (j0 + nt * 16) * NSEQ + k, NSEQ);
        acc = wmma16(af, bf, acc);
      }
#pragma unroll
      for (int r = 0; r < 8; ++r) {
        int i = i0 + mt * 16 + r + 8 * grp;
        int j = j0 + nt * 16 + nn;
        upd[((long)i * NSEQ + j) * CZ + c] = (_Float16)acc[r];
      }
    }
}

// z += sigmoid(LN(z[..])@Wg+bg) * (X@Wo+bo).  readT: operate on z^T positions.
__global__ void k_combine_z(float* z, const _Float16* X,
                            const float* ng, const float* nb,
                            const _Float16* Wg, const float* bg,
                            const _Float16* Wo, const float* bo, int readT) {
  __shared__ _Float16 zn[32 * CZ];
  int p0 = blockIdx.x * 32;
  int rRow = p0 >> 8, c0 = p0 & 255;
  const float* base = readT ? z + ((long)c0 * NSEQ + rRow) * CZ
                            : z + (long)p0 * CZ;
  long stride = readT ? (long)NSEQ * CZ : CZ;
  ln_to_lds(base, stride, CZ, 32, ng, nb, zn);
  __syncthreads();
  int wid = threadIdx.x >> 5, lane = threadIdx.x & 31;
  int nn = lane & 15, grp = lane >> 4;
  int ct = wid;
  for (int rt = 0; rt < 2; ++rt) {
    v8f ag = {}, ay = {};
#pragma unroll
    for (int kk = 0; kk < 4; ++kk) {
      v16h zf = frag_a_rm(zn + rt * 16 * CZ + kk * 32, CZ);
      ag = wmma16(zf, frag_bw(Wg, 4, kk, ct), ag);
      v16h xf = frag_a_rm(X + (long)(p0 + rt * 16) * CZ + kk * 32, CZ);
      ay = wmma16(xf, frag_bw(Wo, 4, kk, ct), ay);
    }
#pragma unroll
    for (int r = 0; r < 8; ++r) {
      int q = rt * 16 + r + 8 * grp, ch = ct * 16 + nn;
      float dv = sigm(ag[r] + bg[ch]) * (ay[r] + bo[ch]);
      long addr = readT ? (((long)(c0 + q)) * NSEQ + rRow) * CZ + ch
                        : ((long)(p0 + q)) * CZ + ch;
      z[addr] += dv;
    }
  }
}

// ---------------------------------------------------------------------------
// triangle attention
// ---------------------------------------------------------------------------
// qkv projection: q/k stored [n][h][i][d]; v stored transposed [n][h][d][i].
__global__ void k_triattn_qkv(const float* z, const float* ng, const float* nb,
                              const _Float16* Wqkv, const float* bqkv,
                              _Float16* qb, _Float16* kb, _Float16* vb,
                              int readT) {
  __shared__ _Float16 zn[32 * CZ];
  int p0 = blockIdx.x * 32;
  int iRow = p0 >> 8, n0 = p0 & 255;
  const float* base = readT ? z + ((long)n0 * NSEQ + iRow) * CZ
                            : z + (long)p0 * CZ;
  long stride = readT ? (long)NSEQ * CZ : CZ;
  ln_to_lds(base, stride, CZ, 32, ng, nb, zn);
  __syncthreads();
  int wid = threadIdx.x >> 5, lane = threadIdx.x & 31;
  int nn = lane & 15, grp = lane >> 4;
  for (int ct = wid; ct < 24; ct += 8)          // N = 384
    for (int rt = 0; rt < 2; ++rt) {
      v8f acc = {};
#pragma unroll
      for (int kk = 0; kk < 4; ++kk) {
        v16h af = frag_a_rm(zn + rt * 16 * CZ + kk * 32, CZ);
        acc = wmma16(af, frag_bw(Wqkv, 4, kk, ct), acc);
      }
      int col = ct * 16 + nn;
      int sec = col >> 7, rem = col & 127, h = rem >> 5, d = rem & 31;
#pragma unroll
      for (int r = 0; r < 8; ++r) {
        int q = rt * 16 + r + 8 * grp;
        int n = n0 + q;
        _Float16 val = (_Float16)(acc[r] + bqkv[col]);
        if (sec == 0)
          qb[(((long)n * 4 + h) * NSEQ + iRow) * 32 + d] = val;
        else if (sec == 1)
          kb[(((long)n * 4 + h) * NSEQ + iRow) * 32 + d] = val;
        else
          vb[(((long)n * 4 + h) * 32 + d) * NSEQ + iRow] = val;
      }
    }
}

// attention core for one (n, h, 64-row i-tile); out -> xout[i][n][c] f16.
__global__ void k_triattn_core(const _Float16* qb, const _Float16* kb,
                               const _Float16* vb, _Float16* xout) {
  __shared__ _Float16 sl[64 * NSEQ];            // 32 KB logits/probs
  int n = blockIdx.z, h = blockIdx.y, i0 = blockIdx.x * 64;
  long nh = (long)n * 4 + h;
  const _Float16* Q = qb + nh * NSEQ * 32;
  const _Float16* K = kb + nh * NSEQ * 32;
  const _Float16* V = vb + nh * 32 * NSEQ;      // transposed [d][j]
  int wid = threadIdx.x >> 5, lane = threadIdx.x & 31;
  int nn = lane & 15, grp = lane >> 4;
  // S = Q K^T / sqrt(32)
  {
    int rt = wid & 3, jh = wid >> 2;
    v16h af = frag_a_rm(Q + (i0 + rt * 16) * 32, 32);
#pragma unroll
    for (int nt = 0; nt < 8; ++nt) {
      v8f acc = {};
      acc = wmma16(af, frag_b_nk(K + (jh * 128 + nt * 16) * 32, 32), acc);
      int j0 = jh * 128 + nt * 16;
#pragma unroll
      for (int r = 0; r < 8; ++r)
        sl[(rt * 16 + r + 8 * grp) * NSEQ + j0 + nn] =
            (_Float16)(acc[r] * 0.17677669529663687f);
    }
  }
  __syncthreads();
  // softmax over j (256) — one wave per 8 rows
  for (int row = wid; row < 64; row += 8) {
    float vals[8];
#pragma unroll
    for (int t = 0; t < 8; ++t) vals[t] = (float)sl[row * NSEQ + lane + 32 * t];
    float mx = vals[0];
#pragma unroll
    for (int t = 1; t < 8; ++t) mx = fmaxf(mx, vals[t]);
    mx = wred_max(mx);
    float sum = 0.f;
#pragma unroll
    for (int t = 0; t < 8; ++t) { vals[t] = __expf(vals[t] - mx); sum += vals[t]; }
    sum = wred_sum(sum);
    float inv = 1.f / sum;
#pragma unroll
    for (int t = 0; t < 8; ++t)
      sl[row * NSEQ + lane + 32 * t] = (_Float16)(vals[t] * inv);
  }
  __syncthreads();
  // O = P V
  {
    int mt = wid & 3, dt = wid >> 2;             // dt in 0..1 (32 = 2 tiles)
    v8f acc = {};
#pragma unroll
    for (int kk = 0; kk < 8; ++kk) {
      v16h af = frag_a_rm(sl + mt * 16 * NSEQ + kk * 32, NSEQ);
      acc = wmma16(af, frag_b_nk(V + (long)(dt * 16) * NSEQ + kk * 32, NSEQ),
                   acc);
    }
#pragma unroll
    for (int r = 0; r < 8; ++r) {
      int i = i0 + mt * 16 + r + 8 * grp;
      int c = h * 32 + dt * 16 + nn;
      xout[((long)i * NSEQ + n) * CZ + c] = (_Float16)acc[r];
    }
  }
}

// ---------------------------------------------------------------------------
// transition (z): x += W2 @ silu(W1 @ LN(x)) fused in-block
// ---------------------------------------------------------------------------
__global__ void k_transition_z(float* z, const float* ng, const float* nb,
                               const _Float16* W1, const float* b1,
                               const _Float16* W2, const float* b2) {
  __shared__ _Float16 zn[32 * CZ];               // 8 KB
  __shared__ _Float16 hid[32 * 512];             // 32 KB
  int p0 = blockIdx.x * 32;
  ln_to_lds(z + (long)p0 * CZ, CZ, CZ, 32, ng, nb, zn);
  __syncthreads();
  int wid = threadIdx.x >> 5, lane = threadIdx.x & 31;
  int nn = lane & 15, grp = lane >> 4;
  for (int ct = wid; ct < 32; ct += 8)
    for (int rt = 0; rt < 2; ++rt) {
      v8f acc = {};
#pragma unroll
      for (int kk = 0; kk < 4; ++kk) {
        v16h af = frag_a_rm(zn + rt * 16 * CZ + kk * 32, CZ);
        acc = wmma16(af, frag_bw(W1, 4, kk, ct), acc);
      }
      int ch = ct * 16 + nn;
#pragma unroll
      for (int r = 0; r < 8; ++r) {
        float v = acc[r] + b1[ch];
        hid[(rt * 16 + r + 8 * grp) * 512 + ch] = (_Float16)(v * sigm(v));
      }
    }
  __syncthreads();
  {
    int ct = wid;
    for (int rt = 0; rt < 2; ++rt) {
      v8f acc = {};
#pragma unroll
      for (int kk = 0; kk < 16; ++kk) {
        v16h af = frag_a_rm(hid + rt * 16 * 512 + kk * 32, 512);
        acc = wmma16(af, frag_bw(W2, 16, kk, ct), acc);
      }
      int ch = ct * 16 + nn;
#pragma unroll
      for (int r = 0; r < 8; ++r) {
        int q = rt * 16 + r + 8 * grp;
        z[((long)(p0 + q)) * CZ + ch] += acc[r] + b2[ch];
      }
    }
  }
}

// ---------------------------------------------------------------------------
// pair-bias attention on s
// ---------------------------------------------------------------------------
// q/k stored [h][i][d] (hd=96); v stored transposed [h][d][i].
__global__ void k_s_qkv(const float* s, const float* nsg, const float* nsb,
                        const _Float16* Wqkv, const float* bqkv,
                        _Float16* qs, _Float16* ks, _Float16* vs,
                        _Float16* snbuf) {
  __shared__ _Float16 sn[32 * CS];               // 24 KB
  int p0 = blockIdx.x * 32;
  ln_to_lds(s + (long)p0 * CS, CS, CS, 32, nsg, nsb, sn);
  __syncthreads();
  for (int idx = threadIdx.x; idx < 32 * CS; idx += blockDim.x)
    snbuf[(long)p0 * CS + idx] = sn[idx];
  int wid = threadIdx.x >> 5, lane = threadIdx.x & 31;
  int nn = lane & 15, grp = lane >> 4;
  for (int ct = wid; ct < 72; ct += 8)           // N = 1152
    for (int rt = 0; rt < 2; ++rt) {
      v8f acc = {};
#pragma unroll
      for (int kk = 0; kk < 12; ++kk) {
        v16h af = frag_a_rm(sn + rt * 16 * CS + kk * 32, CS);
        acc = wmma16(af, frag_bw(Wqkv, 12, kk, ct), acc);
      }
      int col = ct * 16 + nn;
      int sec = col / CS, rem = col % CS, h = rem / 96, d = rem % 96;
#pragma unroll
      for (int r = 0; r < 8; ++r) {
        int i = p0 + rt * 16 + r + 8 * grp;
        _Float16 val = (_Float16)(acc[r] + bqkv[col]);
        if (sec == 0)
          qs[((long)h * NSEQ + i) * 96 + d] = val;
        else if (sec == 1)
          ks[((long)h * NSEQ + i) * 96 + d] = val;
        else
          vs[((long)h * 96 + d) * NSEQ + i] = val;
      }
    }
}

// bias[h][i][j] = LN(z) @ Wbias (padded 128x16) + b
__global__ void k_pair_bias(const float* z, const float* nzg, const float* nzb,
                            const _Float16* Wb16, const float* bb,
                            float* biasb) {
  __shared__ _Float16 zn[32 * CZ];
  int p0 = blockIdx.x * 32;
  ln_to_lds(z + (long)p0 * CZ, CZ, CZ, 32, nzg, nzb, zn);
  __syncthreads();
  int wid = threadIdx.x >> 5, lane = threadIdx.x & 31;
  int nn = lane & 15, grp = lane >> 4;
  if (wid < 2) {
    int rt = wid;
    v8f acc = {};
#pragma unroll
    for (int kk = 0; kk < 4; ++kk) {
      v16h af = frag_a_rm(zn + rt * 16 * CZ + kk * 32, CZ);
      acc = wmma16(af, frag_bw(Wb16, 4, kk, 0), acc);
    }
    if (nn < 4) {
#pragma unroll
      for (int r = 0; r < 8; ++r) {
        int q = rt * 16 + r + 8 * grp;
        biasb[(long)nn * NPOS + p0 + q] = acc[r] + bb[nn];
      }
    }
  }
}

__global__ void k_s_attn(const _Float16* qs, const _Float16* ks,
                         const _Float16* vs, const float* biasb,
                         _Float16* sout) {
  __shared__ _Float16 sl[64 * NSEQ];             // 32 KB
  int h = blockIdx.y, i0 = blockIdx.x * 64;
  const _Float16* Q = qs + (long)h * NSEQ * 96;
  const _Float16* K = ks + (long)h * NSEQ * 96;
  const _Float16* V = vs + (long)h * 96 * NSEQ;  // transposed [d][j]
  int wid = threadIdx.x >> 5, lane = threadIdx.x & 31;
  int nn = lane & 15, grp = lane >> 4;
  {
    int rt = wid & 3, jh = wid >> 2;
#pragma unroll
    for (int nt = 0; nt < 8; ++nt) {
      v8f acc = {};
#pragma unroll
      for (int kk = 0; kk < 3; ++kk) {           // K = 96
        v16h af = frag_a_rm(Q + (i0 + rt * 16) * 96 + kk * 32, 96);
        acc = wmma16(af, frag_b_nk(K + (jh * 128 + nt * 16) * 96 + kk * 32, 96),
                     acc);
      }
      int j0 = jh * 128 + nt * 16;
#pragma unroll
      for (int r = 0; r < 8; ++r) {
        int i = i0 + rt * 16 + r + 8 * grp, j = j0 + nn;
        float v = acc[r] * 0.10206207261596575f +
                  biasb[(long)h * NPOS + (long)i * NSEQ + j];
        sl[(rt * 16 + r + 8 * grp) * NSEQ + j] = (_Float16)v;
      }
    }
  }
  __syncthreads();
  for (int row = wid; row < 64; row += 8) {
    float vals[8];
#pragma unroll
    for (int t = 0; t < 8; ++t) vals[t] = (float)sl[row * NSEQ + lane + 32 * t];
    float mx = vals[0];
#pragma unroll
    for (int t = 1; t < 8; ++t) mx = fmaxf(mx, vals[t]);
    mx = wred_max(mx);
    float sum = 0.f;
#pragma unroll
    for (int t = 0; t < 8; ++t) { vals[t] = __expf(vals[t] - mx); sum += vals[t]; }
    sum = wred_sum(sum);
    float inv = 1.f / sum;
#pragma unroll
    for (int t = 0; t < 8; ++t)
      sl[row * NSEQ + lane + 32 * t] = (_Float16)(vals[t] * inv);
  }
  __syncthreads();
  for (int t = 0; t < 3; ++t) {                  // 24 tiles: 4 mt x 6 dt
    int id = wid + 8 * t, mt = id & 3, dt = id >> 2;
    v8f acc = {};
#pragma unroll
    for (int kk = 0; kk < 8; ++kk) {
      v16h af = frag_a_rm(sl + mt * 16 * NSEQ + kk * 32, NSEQ);
      acc = wmma16(af, frag_b_nk(V + (long)(dt * 16) * NSEQ + kk * 32, NSEQ),
                   acc);
    }
#pragma unroll
    for (int r = 0; r < 8; ++r) {
      int i = i0 + mt * 16 + r + 8 * grp;
      sout[(long)i * CS + h * 96 + dt * 16 + nn] = (_Float16)acc[r];
    }
  }
}

__global__ void k_s_combine(float* s, const _Float16* snbuf,
                            const _Float16* sout,
                            const _Float16* Wg, const float* bg,
                            const _Float16* Wo, const float* bo) {
  int p0 = blockIdx.x * 32;
  int wid = threadIdx.x >> 5, lane = threadIdx.x & 31;
  int nn = lane & 15, grp = lane >> 4;
  for (int ct = wid; ct < 24; ct += 8)
    for (int rt = 0; rt < 2; ++rt) {
      v8f ag = {}, ay = {};
#pragma unroll
      for (int kk = 0; kk < 12; ++kk) {
        v16h gf = frag_a_rm(snbuf + (long)(p0 + rt * 16) * CS + kk * 32, CS);
        ag = wmma16(gf, frag_bw(Wg, 12, kk, ct), ag);
        v16h xf = frag_a_rm(sout + (long)(p0 + rt * 16) * CS + kk * 32, CS);
        ay = wmma16(xf, frag_bw(Wo, 12, kk, ct), ay);
      }
      int ch = ct * 16 + nn;
#pragma unroll
      for (int r = 0; r < 8; ++r) {
        int q = rt * 16 + r + 8 * grp;
        s[((long)(p0 + q)) * CS + ch] +=
            sigm(ag[r] + bg[ch]) * (ay[r] + bo[ch]);
      }
    }
}

// ---------------------------------------------------------------------------
// transition (s): two-pass via hbuf (256 x 1536 f16)
// ---------------------------------------------------------------------------
__global__ void k_trans_s_1(const float* s, const float* ng, const float* nb,
                            const _Float16* W1, const float* b1,
                            _Float16* hbuf) {
  __shared__ _Float16 xn[16 * CS];               // 12 KB
  int i0 = blockIdx.x * 16;
  ln_to_lds(s + (long)i0 * CS, CS, CS, 16, ng, nb, xn);
  __syncthreads();
  int wid = threadIdx.x >> 5, lane = threadIdx.x & 31;
  int nn = lane & 15, grp = lane >> 4;
  for (int ct = wid; ct < 96; ct += 8) {         // N = 1536
    v8f acc = {};
#pragma unroll
    for (int kk = 0; kk < 12; ++kk) {
      v16h af = frag_a_rm(xn + kk * 32, CS);
      acc = wmma16(af, frag_bw(W1, 12, kk, ct), acc);
    }
    int ch = ct * 16 + nn;
#pragma unroll
    for (int r = 0; r < 8; ++r) {
      float v = acc[r] + b1[ch];
      hbuf[(long)(i0 + r + 8 * grp) * 1536 + ch] = (_Float16)(v * sigm(v));
    }
  }
}
__global__ void k_trans_s_2(float* s, const _Float16* hbuf,
                            const _Float16* W2, const float* b2) {
  int i0 = blockIdx.x * 16;
  int wid = threadIdx.x >> 5, lane = threadIdx.x & 31;
  int nn = lane & 15, grp = lane >> 4;
  for (int ct = wid; ct < 24; ct += 8) {         // N = 384
    v8f acc = {};
#pragma unroll
    for (int kk = 0; kk < 48; ++kk) {            // K = 1536
      v16h af = frag_a_rm(hbuf + (long)i0 * 1536 + kk * 32, 1536);
      acc = wmma16(af, frag_bw(W2, 48, kk, ct), acc);
    }
    int ch = ct * 16 + nn;
#pragma unroll
    for (int r = 0; r < 8; ++r)
      s[(long)(i0 + r + 8 * grp) * CS + ch] += acc[r] + b2[ch];
  }
}

// ---------------------------------------------------------------------------
// host orchestration
// ---------------------------------------------------------------------------
extern "C" void kernel_launch(void* const* d_in, const int* in_sizes, int n_in,
                              void* d_out, int out_size, void* d_ws,
                              size_t ws_size, hipStream_t stream) {
  (void)in_sizes; (void)n_in; (void)out_size; (void)ws_size;
  // params flattened in jax pytree order (dict keys sorted):
  //  0 s, 1 z, then apb, tae, tas, tmi, tmo, ts, tz (each dict key-sorted,
  //  Linear = (W, b)).
  auto IN = [&](int i) { return (const float*)d_in[i]; };

  float* outS = (float*)d_out;            // running s (256 x 384)
  float* outZ = outS + NSEQ * CS;         // running z (256 x 256 x 128)

  // ---- workspace carve-out ----
  size_t cur = 0;
  auto takeH = [&](size_t elems) {
    _Float16* p = (_Float16*)((char*)d_ws + cur);
    cur = (cur + elems * sizeof(_Float16) + 255) & ~(size_t)255;
    return p;
  };
  auto takeF = [&](size_t elems) {
    float* p = (float*)((char*)d_ws + cur);
    cur = (cur + elems * sizeof(float) + 255) & ~(size_t)255;
    return p;
  };
  // f16 swizzled weights
  _Float16 *tmo_pa = takeH(16384), *tmo_ga = takeH(16384), *tmo_pb = takeH(16384),
           *tmo_gb = takeH(16384), *tmo_og = takeH(16384), *tmo_op = takeH(16384);
  _Float16 *tmi_pa = takeH(16384), *tmi_ga = takeH(16384), *tmi_pb = takeH(16384),
           *tmi_gb = takeH(16384), *tmi_og = takeH(16384), *tmi_op = takeH(16384);
  _Float16 *tas_qkv = takeH(49152), *tas_gate = takeH(16384), *tas_op = takeH(16384);
  _Float16 *tae_qkv = takeH(49152), *tae_gate = takeH(16384), *tae_op = takeH(16384);
  _Float16 *tz_w1 = takeH(65536), *tz_w2 = takeH(65536);
  _Float16 *apb_qkv = takeH(442368), *apb_gate = takeH(147456),
           *apb_op = takeH(147456), *apb_bias = takeH(2048);
  _Float16 *ts_w1 = takeH(589824), *ts_w2 = takeH(589824);
  // big activation buffers (f16). a_t/b_t reused as q/k buffers in attention.
  _Float16* a_t  = takeH((size_t)CZ * NPOS);
  _Float16* b_t  = takeH((size_t)CZ * NPOS);
  _Float16* vbuf = takeH((size_t)CZ * NPOS);
  _Float16* xbuf = takeH((size_t)CZ * NPOS);
  float*    biasb = takeF((size_t)4 * NPOS);
  _Float16 *qs = takeH(98304), *ks = takeH(98304), *vs = takeH(98304);
  _Float16 *snbuf = takeH(98304), *soutbuf = takeH(98304);
  _Float16* hbuf = takeH((size_t)NSEQ * 1536);

  // ---- init running s/z in d_out ----
  hipMemcpyAsync(outS, d_in[0], (size_t)NSEQ * CS * sizeof(float),
                 hipMemcpyDeviceToDevice, stream);
  hipMemcpyAsync(outZ, d_in[1], (size_t)NPOS * CZ * sizeof(float),
                 hipMemcpyDeviceToDevice, stream);

  // ---- weight conversion + swizzle ----
  auto cvtw = [&](const float* src, _Float16* dst, int K, int N, int srcN) {
    int n = K * N;
    int g = (n + 255) / 256;
    k_cvt_swz<<<dim3(g), dim3(256), 0, stream>>>(src, dst, K, N, srcN);
  };
  cvtw(IN(54), tmo_pa, 128, 128, 128); cvtw(IN(44), tmo_ga, 128, 128, 128);
  cvtw(IN(56), tmo_pb, 128, 128, 128); cvtw(IN(46), tmo_gb, 128, 128, 128);
  cvtw(IN(50), tmo_og, 128, 128, 128); cvtw(IN(52), tmo_op, 128, 128, 128);
  cvtw(IN(40), tmi_pa, 128, 128, 128); cvtw(IN(30), tmi_ga, 128, 128, 128);
  cvtw(IN(42), tmi_pb, 128, 128, 128); cvtw(IN(32), tmi_gb, 128, 128, 128);
  cvtw(IN(36), tmi_og, 128, 128, 128); cvtw(IN(38), tmi_op, 128, 128, 128);
  cvtw(IN(28), tas_qkv, 128, 384, 384);
  cvtw(IN(22), tas_gate, 128, 128, 128); cvtw(IN(26), tas_op, 128, 128, 128);
  cvtw(IN(20), tae_qkv, 128, 384, 384);
  cvtw(IN(14), tae_gate, 128, 128, 128); cvtw(IN(18), tae_op, 128, 128, 128);
  cvtw(IN(66), tz_w1, 128, 512, 512);  cvtw(IN(68), tz_w2, 512, 128, 128);
  cvtw(IN(12), apb_qkv, 384, 1152, 1152);
  cvtw(IN(4),  apb_gate, 384, 384, 384); cvtw(IN(10), apb_op, 384, 384, 384);
  cvtw(IN(2),  apb_bias, 128, 16, 4);
  cvtw(IN(60), ts_w1, 384, 1536, 1536); cvtw(IN(62), ts_w2, 1536, 384, 384);

  dim3 B256(256);
  dim3 gPos(NPOS / 32);           // 2048 position-tile blocks

  // ---- tri_mul outgoing (tmo: ga44 gb46 nb48 ng49 og50 op52 pa54 pb56) ----
  k_trimul_proj<<<gPos, B256, 0, stream>>>(outZ, IN(49), IN(48),
      tmo_pa, IN(55), tmo_ga, IN(45), tmo_pb, IN(57), tmo_gb, IN(47),
      a_t, b_t, 0);
  k_trimul_einsum<<<dim3(4, 4, 16), B256, 0, stream>>>(a_t, b_t, xbuf);
  k_combine_z<<<gPos, B256, 0, stream>>>(outZ, xbuf, IN(49), IN(48),
      tmo_og, IN(51), tmo_op, IN(53), 0);

  // ---- tri_mul incoming (tmi: 30..43), transposed a/b layouts ----
  k_trimul_proj<<<gPos, B256, 0, stream>>>(outZ, IN(35), IN(34),
      tmi_pa, IN(41), tmi_ga, IN(31), tmi_pb, IN(43), tmi_gb, IN(33),
      a_t, b_t, 1);
  k_trimul_einsum<<<dim3(4, 4, 16), B256, 0, stream>>>(a_t, b_t, xbuf);
  k_combine_z<<<gPos, B256, 0, stream>>>(outZ, xbuf, IN(35), IN(34),
      tmi_og, IN(37), tmi_op, IN(39), 0);

  // ---- tri_attn starting (tas: gate22 nb24 ng25 op26 qkv28) ----
  k_triattn_qkv<<<gPos, B256, 0, stream>>>(outZ, IN(25), IN(24),
      tas_qkv, IN(29), a_t, b_t, vbuf, 0);
  k_triattn_core<<<dim3(4, 4, NSEQ), B256, 0, stream>>>(a_t, b_t, vbuf, xbuf);
  k_combine_z<<<gPos, B256, 0, stream>>>(outZ, xbuf, IN(25), IN(24),
      tas_gate, IN(23), tas_op, IN(27), 0);

  // ---- tri_attn ending (tae: gate14 nb16 ng17 op18 qkv20), transposed ----
  k_triattn_qkv<<<gPos, B256, 0, stream>>>(outZ, IN(17), IN(16),
      tae_qkv, IN(21), a_t, b_t, vbuf, 1);
  k_triattn_core<<<dim3(4, 4, NSEQ), B256, 0, stream>>>(a_t, b_t, vbuf, xbuf);
  k_combine_z<<<gPos, B256, 0, stream>>>(outZ, xbuf, IN(17), IN(16),
      tae_gate, IN(15), tae_op, IN(19), 1);

  // ---- transition z (tz: nb64 ng65 w1 66 w2 68) ----
  k_transition_z<<<gPos, B256, 0, stream>>>(outZ, IN(65), IN(64),
      tz_w1, IN(67), tz_w2, IN(69));

  // ---- pair-bias attention on s (apb) ----
  k_s_qkv<<<dim3(8), B256, 0, stream>>>(outS, IN(7), IN(6),
      apb_qkv, IN(13), qs, ks, vs, snbuf);
  k_pair_bias<<<gPos, B256, 0, stream>>>(outZ, IN(9), IN(8),
      apb_bias, IN(3), biasb);
  k_s_attn<<<dim3(4, 4), B256, 0, stream>>>(qs, ks, vs, biasb, soutbuf);
  k_s_combine<<<dim3(8), B256, 0, stream>>>(outS, snbuf, soutbuf,
      apb_gate, IN(5), apb_op, IN(11));

  // ---- transition s (ts: nb58 ng59 w1 60 w2 62) ----
  k_trans_s_1<<<dim3(16), B256, 0, stream>>>(outS, IN(59), IN(58),
      ts_w1, IN(61), hbuf);
  k_trans_s_2<<<dim3(16), B256, 0, stream>>>(outS, hbuf, ts_w2, IN(63));
}
